// RotAttentionPool2d_43696997270172
// MI455X (gfx1250) — matrix operand order repaired and checked by reading, not compiled, not run on hardware
//
#include <hip/hip_runtime.h>
#include <hip/hip_bf16.h>

// ---------------------------------------------------------------------------
// RotAttentionPool2d, specialized: only out[:, 0] (the mean/CLS token) is
// needed. B=64, C=E=1024, H=W=16 (n=256 tokens + 1 mean token), 8 heads x 128.
//
// Heavy work = K/V GEMM (16448 x 1024 x 2048, ~69 GFLOP) -> f16 WMMA w/ f32
// accumulate, fed from double-buffered LDS tiles filled with async
// global->LDS copies (ASYNCcnt path) so WMMA never waits on HBM latency.
// ---------------------------------------------------------------------------

typedef _Float16 v16h __attribute__((ext_vector_type(16)));
typedef float    v8f  __attribute__((ext_vector_type(8)));

union HalfFrag { v16h v; uint4 q[2]; };

#define B_     64
#define C_     1024
#define NTOK   257          // 256 spatial + 1 mean token
#define RPAD   272          // 17 * 16, padded token rows per batch
#define KC     64           // K chunk staged per LDS buffer
#define LDSTR  72           // padded LDS row stride in halfs (64 data + 8 pad)

// ---------------------------------------------------------------------------
// 16-byte global -> LDS copy: async (ASYNCcnt, bypasses VGPRs) when the
// gfx1250 builtin is available, else plain load + ds_store.
// ---------------------------------------------------------------------------
#if defined(__gfx1250__) && __has_builtin(__builtin_amdgcn_global_load_async_to_lds_b128)
#define ASYNC_LDS_COPY 1
#else
#define ASYNC_LDS_COPY 0
#endif

#if ASYNC_LDS_COPY
typedef int v4i_gs __attribute__((vector_size(16)));     // generic-vector int4
typedef __attribute__((address_space(1))) v4i_gs* g_v4i_ptr;
typedef __attribute__((address_space(3))) v4i_gs* l_v4i_ptr;
#endif

__device__ __forceinline__ void cp16_g2l(const _Float16* g, _Float16* l) {
#if ASYNC_LDS_COPY
  // generic->AS1 keeps the 64-bit address; generic->AS3 is the low 32 bits
  // (ISA 10.2: the LDS address is the low 32 bits within the shared aperture).
  __builtin_amdgcn_global_load_async_to_lds_b128(
      (g_v4i_ptr)(unsigned long long)g,
      (l_v4i_ptr)(unsigned int)(unsigned long long)l,
      0, 0);
#else
  *(uint4*)l = *(const uint4*)g;
#endif
}

__device__ __forceinline__ void cp_wait_all() {
#if ASYNC_LDS_COPY
#if __has_builtin(__builtin_amdgcn_s_wait_asynccnt)
  __builtin_amdgcn_s_wait_asynccnt(0);
#else
  asm volatile("s_wait_asynccnt 0x0" ::: "memory");
#endif
#endif
}

// ---------------------------------------------------------------------------
// 1) Transpose x (B, C, 256) f32 -> tA (B, RPAD, C) f16 at token rows 1..256
// ---------------------------------------------------------------------------
__global__ __launch_bounds__(256) void transpose_pack_kernel(
    const float* __restrict__ x, _Float16* __restrict__ tA) {
  const int b    = blockIdx.z;
  const int cblk = blockIdx.y;   // 32 blocks of 32 channels
  const int jblk = blockIdx.x;   // 8 blocks of 32 tokens
  __shared__ float tile[32][33];
  const int tid = threadIdx.x;
  const int jl  = tid & 31;
  const int cq  = tid >> 5;      // 0..7
  const int cbase = cblk * 32, jbase = jblk * 32;
#pragma unroll
  for (int r = 0; r < 4; ++r) {
    const int cl = cq * 4 + r;
    tile[cl][jl] = x[((size_t)b * C_ + cbase + cl) * 256 + jbase + jl];
  }
  __syncthreads();
#pragma unroll
  for (int r = 0; r < 4; ++r) {
    const int jl2 = cq * 4 + r;
    const int cl2 = jl;
    tA[((size_t)b * RPAD + 1 + jbase + jl2) * C_ + cbase + cl2] =
        (_Float16)tile[cl2][jl2];
  }
}

// ---------------------------------------------------------------------------
// 2) Mean token: tA[b, 0, c] = mean_j x[b, c, j]   (one wave per channel)
// ---------------------------------------------------------------------------
__global__ __launch_bounds__(256) void mean_pack_kernel(
    const float* __restrict__ x, _Float16* __restrict__ tA) {
  const int b    = blockIdx.x;
  const int wid  = threadIdx.x >> 5;
  const int lane = threadIdx.x & 31;
  const int c    = blockIdx.y * 8 + wid;
  const float* xr = x + ((size_t)b * C_ + c) * 256;
  float s = 0.f;
#pragma unroll
  for (int k = 0; k < 8; ++k) s += xr[lane + 32 * k];
#pragma unroll
  for (int off = 16; off > 0; off >>= 1) s += __shfl_down(s, off, 32);
  if (lane == 0) tA[(size_t)b * RPAD * C_ + c] = (_Float16)(s * (1.f / 256.f));
}

// ---------------------------------------------------------------------------
// 3) Convert qkv_w rows [1024, 3072) (K and V weights) f32 -> f16
// ---------------------------------------------------------------------------
__global__ __launch_bounds__(256) void convert_w_kernel(
    const float* __restrict__ qkv_w, _Float16* __restrict__ Wkv) {
  const int idx = blockIdx.x * 256 + threadIdx.x;    // each handles 4 floats
  const float4 w = *(const float4*)(qkv_w + (size_t)1024 * 1024 + (size_t)idx * 4);
  union { _Float16 h[4]; uint2 u; } pk;
  pk.h[0] = (_Float16)w.x; pk.h[1] = (_Float16)w.y;
  pk.h[2] = (_Float16)w.z; pk.h[3] = (_Float16)w.w;
  *(uint2*)(Wkv + (size_t)idx * 4) = pk.u;
}

// ---------------------------------------------------------------------------
// 4) RoPE sin/cos tables: 256 positions x 64 angle pairs
// ---------------------------------------------------------------------------
__global__ __launch_bounds__(256) void sincos_kernel(
    float* __restrict__ sintab, float* __restrict__ costab) {
  const int idx = blockIdx.x * 256 + threadIdx.x;    // 16384 entries
  const int p = idx >> 6, i = idx & 63;
  const int y = p >> 4, xg = p & 15;
  const float coord = (i < 32) ? (-1.f + 2.f * (float)y / 15.f)
                               : (-1.f + 2.f * (float)xg / 15.f);
  // bands = pi * 2^(k*(log2(224)-1)/31), k = i%32
  const float step = (log2f(224.f) - 1.f) / 31.f;
  const float band = 3.14159265358979323846f * exp2f((float)(i & 31) * step);
  const float ang = coord * band;
  sintab[idx] = sinf(ang);
  costab[idx] = cosf(ang);
}

// ---------------------------------------------------------------------------
// 5) WMMA GEMM: KV[b, part, row, feat] = tA[b] @ Wkv^T + bias
//    block tile 64(M) x 128(N), 8 waves (wave tile 16 x 64, 4 frags),
//    K chunks of 64 double-buffered in LDS via async global->LDS copies.
// ---------------------------------------------------------------------------
__global__ __launch_bounds__(256) void gemm_kv_kernel(
    const _Float16* __restrict__ tA, const _Float16* __restrict__ Wkv,
    const float* __restrict__ qkv_b, _Float16* __restrict__ KV) {
  const int b    = blockIdx.z;
  const int mblk = blockIdx.y;            // 0..4 -> 64 rows each (rows 0..319)
  const int nblk = blockIdx.x;            // 0..15 -> 128 cols each
  const int tid  = threadIdx.x;
  const int lane = tid & 31;
  const int wid  = tid >> 5;              // 0..7
  const int wave_m = wid & 3;             // 4 M positions (16 rows each)
  const int wave_n = wid >> 2;            // 2 N positions (64 cols each)

  __shared__ _Float16 Asm[2][64][LDSTR];
  __shared__ _Float16 Bsm[2][128][LDSTR];

  // -------- staging: A = 64x64 halfs (512 x 16B), B = 128x64 (1024 x 16B) ---
  const _Float16* tAb  = tA  + (size_t)b * RPAD * C_;
  const _Float16* Wb   = Wkv + (size_t)nblk * 128 * C_;

  auto stage = [&](int kb, int buf) {
#pragma unroll
    for (int it = 0; it < 2; ++it) {             // A tile
      const int i = tid + it * 256;
      const int row = i >> 3, seg = i & 7;
      int grow = mblk * 64 + row;
      grow = grow < RPAD ? grow : (RPAD - 1);    // clamp; stores are guarded
      cp16_g2l(tAb + (size_t)grow * C_ + kb + seg * 8, &Asm[buf][row][seg * 8]);
    }
#pragma unroll
    for (int it = 0; it < 4; ++it) {             // B tile
      const int i = tid + it * 256;
      const int row = i >> 3, seg = i & 7;
      cp16_g2l(Wb + (size_t)row * C_ + kb + seg * 8, &Bsm[buf][row][seg * 8]);
    }
  };

  v8f acc[4] = {};
  const int arow = wave_m * 16 + (lane & 15);
  const int kg   = lane >> 4;

  stage(0, 0);

  for (int kc = 0; kc < C_ / KC; ++kc) {
    cp_wait_all();          // my async copies into buf[kc&1] are done
    __syncthreads();        // everyone's copies visible / prior reads retired
    if (kc + 1 < C_ / KC) stage((kc + 1) * KC, (kc + 1) & 1);

    const _Float16* ap = &Asm[kc & 1][arow][kg * 8];
#pragma unroll
    for (int ks = 0; ks < 2; ++ks) {             // two 16x16x32 steps per chunk
      HalfFrag a;
      a.q[0] = *(const uint4*)(ap + ks * 32);        // K = ks*32 + kg*8  + 0..7
      a.q[1] = *(const uint4*)(ap + ks * 32 + 16);   // K = ks*32 + 16 + kg*8
#pragma unroll
      for (int nt = 0; nt < 4; ++nt) {
        const _Float16* bp =
            &Bsm[kc & 1][wave_n * 64 + nt * 16 + (lane & 15)][ks * 32 + kg * 8];
        HalfFrag bb;
        bb.q[0] = *(const uint4*)(bp);               // K = ks*32 + kg*8 + 0..7
        bb.q[1] = *(const uint4*)(bp + 16);          // K = ks*32 + 16 + kg*8 + 0..7
        acc[nt] = __builtin_amdgcn_wmma_f32_16x16x32_f16(
            false, a.v, false, bb.v, (short)0, acc[nt], false, false);
      }
    }
    __syncthreads();        // done reading buf[kc&1] before it is refilled
  }

  // --- epilogue: bias + f16 store, guarded to valid 257 token rows ---
  const int row_base = mblk * 64 + wave_m * 16 + (lane >> 4) * 8;
  const int nwave    = nblk * 128 + wave_n * 64;
#pragma unroll
  for (int nt = 0; nt < 4; ++nt) {
    const int n    = nwave + nt * 16 + (lane & 15);
    const float bias = qkv_b[1024 + n];
    const int part = n >> 10;                   // 0 = K, 1 = V
    const int feat = n & 1023;
#pragma unroll
    for (int r = 0; r < 8; ++r) {
      const int row = row_base + r;
      if (row < NTOK) {
        KV[(((size_t)b * 2 + part) * NTOK + row) * C_ + feat] =
            (_Float16)(acc[nt][r] + bias);
      }
    }
  }
}

// ---------------------------------------------------------------------------
// 6) q0 GEMV: q0[b, n] = tA[b,0,:] . qkv_w[n,:] + qkv_b[n]   (n in [0,1024))
// ---------------------------------------------------------------------------
__global__ __launch_bounds__(256) void q0_kernel(
    const _Float16* __restrict__ tA, const float* __restrict__ qkv_w,
    const float* __restrict__ qkv_b, float* __restrict__ q0) {
  const int n = blockIdx.x * 256 + threadIdx.x;
  const int b = blockIdx.y;
  const _Float16* trow = tA + (size_t)b * RPAD * C_;
  const float* wr = qkv_w + (size_t)n * C_;
  float acc = qkv_b[n];
  for (int c = 0; c < C_; ++c) acc += (float)trow[c] * wr[c];
  q0[(size_t)b * C_ + n] = acc;
}

// ---------------------------------------------------------------------------
// 7) One attention row per (b, head): RoPE(K) on the fly, softmax, AV
// ---------------------------------------------------------------------------
__global__ __launch_bounds__(256) void attention_kernel(
    const float* __restrict__ q0, const _Float16* __restrict__ KV,
    const float* __restrict__ sintab, const float* __restrict__ costab,
    float* __restrict__ pooled) {
  const int b = blockIdx.x;
  const int h = blockIdx.y;
  const int tid = threadIdx.x;
  __shared__ float qs[128];
  __shared__ float logits[NTOK];
  __shared__ float red[256];

  if (tid < 128) qs[tid] = q0[(size_t)b * C_ + h * 128 + tid];
  __syncthreads();

  const float scale = 0.08838834764831845f;   // 128^-0.5
  const _Float16* Kb = KV + (size_t)b * 2 * NTOK * C_ + h * 128;
  const _Float16* Vb = Kb + (size_t)NTOK * C_;

  for (int j = tid; j < NTOK; j += 256) {
    const _Float16* kr = Kb + (size_t)j * C_;
    float dot = 0.f;
    if (j == 0) {                // mean token: no RoPE
      for (int d = 0; d < 128; ++d) dot += qs[d] * (float)kr[d];
    } else {
      const int p = j - 1;
      for (int i = 0; i < 64; ++i) {
        const float ci = costab[p * 64 + i], si = sintab[p * 64 + i];
        const float k0 = (float)kr[2 * i], k1 = (float)kr[2 * i + 1];
        dot += qs[2 * i]     * (k0 * ci - k1 * si)
             + qs[2 * i + 1] * (k1 * ci + k0 * si);
      }
    }
    logits[j] = dot * scale;
  }
  __syncthreads();

  float lmax = -1e30f;
  for (int j = tid; j < NTOK; j += 256) lmax = fmaxf(lmax, logits[j]);
  red[tid] = lmax; __syncthreads();
  for (int s = 128; s > 0; s >>= 1) {
    if (tid < s) red[tid] = fmaxf(red[tid], red[tid + s]);
    __syncthreads();
  }
  const float gmax = red[0]; __syncthreads();

  float lsum = 0.f;
  for (int j = tid; j < NTOK; j += 256) {
    const float e = __expf(logits[j] - gmax);
    logits[j] = e;
    lsum += e;
  }
  red[tid] = lsum; __syncthreads();
  for (int s = 128; s > 0; s >>= 1) {
    if (tid < s) red[tid] += red[tid + s];
    __syncthreads();
  }
  const float inv = 1.f / red[0];
  __syncthreads();

  if (tid < 128) {
    float acc = 0.f;
    for (int j = 0; j < NTOK; ++j) acc += logits[j] * (float)Vb[(size_t)j * C_ + tid];
    pooled[(size_t)b * C_ + h * 128 + tid] = acc * inv;
  }
}

// ---------------------------------------------------------------------------
// 8) Projection: out[b, n] = pooled[b,:] . proj_w[n,:] + proj_b[n]
// ---------------------------------------------------------------------------
__global__ __launch_bounds__(256) void proj_kernel(
    const float* __restrict__ pooled, const float* __restrict__ proj_w,
    const float* __restrict__ proj_b, float* __restrict__ out) {
  const int n = blockIdx.x * 256 + threadIdx.x;
  const int b = blockIdx.y;
  const float* pr = pooled + (size_t)b * C_;
  const float* wr = proj_w + (size_t)n * C_;
  float acc = proj_b[n];
  for (int c = 0; c < C_; ++c) acc += pr[c] * wr[c];
  out[(size_t)b * C_ + n] = acc;
}

// ---------------------------------------------------------------------------
// Launch
// ---------------------------------------------------------------------------
extern "C" void kernel_launch(void* const* d_in, const int* in_sizes, int n_in,
                              void* d_out, int out_size, void* d_ws, size_t ws_size,
                              hipStream_t stream) {
  const float* x      = (const float*)d_in[0];
  const float* qkv_w  = (const float*)d_in[1];
  const float* qkv_b  = (const float*)d_in[2];
  const float* proj_w = (const float*)d_in[3];
  const float* proj_b = (const float*)d_in[4];
  float* out = (float*)d_out;

  char* ws = (char*)d_ws;
  // workspace layout (bytes)
  _Float16* tA     = (_Float16*)(ws);                       // 64*272*1024*2 = 35,651,584
  _Float16* Wkv    = (_Float16*)(ws + 35651584);            // 2048*1024*2   =  4,194,304
  _Float16* KV     = (_Float16*)(ws + 39845888);            // 64*2*257*1024*2 = 67,371,008
  float*    q0     = (float*)   (ws + 107216896);           // 64*1024*4     =    262,144
  float*    pooled = (float*)   (ws + 107479040);           // 64*1024*4     =    262,144
  float*    sintab = (float*)   (ws + 107741184);           // 256*64*4      =     65,536
  float*    costab = (float*)   (ws + 107806720);           // 256*64*4      =     65,536
  (void)in_sizes; (void)n_in; (void)out_size; (void)ws_size;

  transpose_pack_kernel<<<dim3(8, 32, B_), 256, 0, stream>>>(x, tA);
  mean_pack_kernel<<<dim3(B_, 128), 256, 0, stream>>>(x, tA);
  convert_w_kernel<<<dim3(2048), 256, 0, stream>>>(qkv_w, Wkv);
  sincos_kernel<<<dim3(64), 256, 0, stream>>>(sintab, costab);
  q0_kernel<<<dim3(4, B_), 256, 0, stream>>>(tA, qkv_w, qkv_b, q0);
  gemm_kv_kernel<<<dim3(16, 5, B_), 256, 0, stream>>>(tA, Wkv, qkv_b, KV);
  attention_kernel<<<dim3(B_, 8), 256, 0, stream>>>(q0, KV, sintab, costab, pooled);
  proj_kernel<<<dim3(4, B_), 256, 0, stream>>>(pooled, proj_w, proj_b, out);
}
